// Decoder_40089224741260
// MI455X (gfx1250) — compile-verified, matrix operand order checked by reading, not run on hardware
//
#include <hip/hip_runtime.h>
#include <math.h>

#define EE 256
#define CCAM 6
#define LNUM 6
#define TQ 900
#define BB 2
#define NCLS 10
#define HH 32
#define WW 88
#define NHH 8
#define HD 32
#define NREAL (BB * TQ)   /* 1800 */
#define NPAD 1856         /* 116 tiles of 16; 29 chunks of 32 per batch: 900+928 <= 1856 */
#define RT (NPAD / 16)    /* 116 row tiles */

typedef __attribute__((ext_vector_type(16))) _Float16 v16h;
typedef __attribute__((ext_vector_type(8)))  float    v8f;

// Load a 16-element f16 WMMA fragment from f32 memory: two 8-float runs at
// p[0..7] and p[16..23], each as 2x b128 loads. All call sites are 32B-aligned.
__device__ __forceinline__ v16h load_frag(const float* __restrict__ p)
{
  const float4 r0 = ((const float4*)p)[0];
  const float4 r1 = ((const float4*)p)[1];
  const float4 r2 = ((const float4*)(p + 16))[0];
  const float4 r3 = ((const float4*)(p + 16))[1];
  v16h f;
  f[0]  = (_Float16)r0.x; f[1]  = (_Float16)r0.y; f[2]  = (_Float16)r0.z; f[3]  = (_Float16)r0.w;
  f[4]  = (_Float16)r1.x; f[5]  = (_Float16)r1.y; f[6]  = (_Float16)r1.z; f[7]  = (_Float16)r1.w;
  f[8]  = (_Float16)r2.x; f[9]  = (_Float16)r2.y; f[10] = (_Float16)r2.z; f[11] = (_Float16)r2.w;
  f[12] = (_Float16)r3.x; f[13] = (_Float16)r3.y; f[14] = (_Float16)r3.z; f[15] = (_Float16)r3.w;
  return f;
}

// ---------------------------------------------------------------------------
// WMMA GEMM: out[n][m] = act( A[n][:] . W[m][:] + bias[m] ) (+ res)
// A: NPAD x K (ld=K), W: M x K row-major, out ld = ldout.
// One wave per 16x64 tile (1 A-frag reused over 4 B-frags -> 4 WMMA / k-step).
// grid = (RT, ceil(M/64)). act: 0 none, 1 gelu(exact), 2 leaky(0.01), 3 sigmoid
// ---------------------------------------------------------------------------
__global__ __launch_bounds__(32) void gemm_kernel(
    const float* __restrict__ A, const float* __restrict__ W,
    const float* __restrict__ bias, const float* __restrict__ res,
    float* __restrict__ out, int K, int M, int ldout, int act)
{
  const int lane = threadIdx.x & 31;
  const int half = lane >> 4;
  const int idx  = lane & 15;
  const int hb   = half * 8;
  const int r0   = blockIdx.x * 16;
  const int cb   = blockIdx.y * 64;

  const float* pa = A + (size_t)(r0 + idx) * K + hb;
  const float* pw[4];
  int col[4], wr[4];
#pragma unroll
  for (int ct = 0; ct < 4; ++ct) {
    col[ct] = cb + ct * 16 + idx;
    wr[ct]  = (col[ct] < M) ? col[ct] : (M - 1);
    pw[ct]  = W + (size_t)wr[ct] * K + hb;
  }

  v8f acc0 = {}, acc1 = {}, acc2 = {}, acc3 = {};
  for (int k0 = 0; k0 < K; k0 += 32) {
    if (k0 + 32 < K) {
      __builtin_prefetch((const void*)(pa + k0 + 32), 0, 1);
      __builtin_prefetch((const void*)(pw[0] + k0 + 32), 0, 1);
      __builtin_prefetch((const void*)(pw[2] + k0 + 32), 0, 1);
    }
    const v16h af = load_frag(pa + k0);
    const v16h b0 = load_frag(pw[0] + k0);
    const v16h b1 = load_frag(pw[1] + k0);
    const v16h b2 = load_frag(pw[2] + k0);
    const v16h b3 = load_frag(pw[3] + k0);
    acc0 = __builtin_amdgcn_wmma_f32_16x16x32_f16(false, af, false, b0, (short)0, acc0, false, false);
    acc1 = __builtin_amdgcn_wmma_f32_16x16x32_f16(false, af, false, b1, (short)0, acc1, false, false);
    acc2 = __builtin_amdgcn_wmma_f32_16x16x32_f16(false, af, false, b2, (short)0, acc2, false, false);
    acc3 = __builtin_amdgcn_wmma_f32_16x16x32_f16(false, af, false, b3, (short)0, acc3, false, false);
  }

#pragma unroll
  for (int ct = 0; ct < 4; ++ct) {
    const v8f& acc = (ct == 0) ? acc0 : (ct == 1) ? acc1 : (ct == 2) ? acc2 : acc3;
    if (col[ct] < M) {
      const float b = bias[wr[ct]];
#pragma unroll
      for (int r = 0; r < 8; ++r) {
        float v = acc[r] + b;
        if (act == 1)      v = 0.5f * v * (1.0f + erff(v * 0.70710678118f));
        else if (act == 2) v = (v > 0.0f) ? v : 0.01f * v;
        else if (act == 3) v = 1.0f / (1.0f + __expf(-v));
        const int row = r0 + r + 8 * half;
        float o = v;
        if (res) o += res[(size_t)row * ldout + col[ct]];
        out[(size_t)row * ldout + col[ct]] = o;
      }
    }
  }
}

// ---------------------------------------------------------------------------
// LayerNorm over E=256, one block per row; optional fused exact GELU.
// ---------------------------------------------------------------------------
__global__ __launch_bounds__(256) void ln_kernel(
    const float* __restrict__ x, const float* __restrict__ s,
    const float* __restrict__ b, float* __restrict__ out, int dogelu)
{
  __shared__ float red[256];
  const int row = blockIdx.x;
  const int e   = threadIdx.x;
  const float v = x[(size_t)row * EE + e];
  red[e] = v;
  __syncthreads();
  for (int o = 128; o > 0; o >>= 1) { if (e < o) red[e] += red[e + o]; __syncthreads(); }
  const float mean = red[0] * (1.0f / EE);
  __syncthreads();
  const float d = v - mean;
  red[e] = d * d;
  __syncthreads();
  for (int o = 128; o > 0; o >>= 1) { if (e < o) red[e] += red[e + o]; __syncthreads(); }
  const float var = red[0] * (1.0f / EE);
  float y = d * rsqrtf(var + 1e-5f) * s[e] + b[e];
  if (dogelu) y = 0.5f * y * (1.0f + erff(y * 0.70710678118f));
  out[(size_t)row * EE + e] = y;
}

// ---------------------------------------------------------------------------
// MHCA projective bilinear sampling. One block per (t,b), thread = channel e.
// Replicates reference quirks: anchor re-affined per camera; pix.x vs H,
// pix.y vs W; grid x -> width axis, grid y -> height axis.
// ---------------------------------------------------------------------------
__global__ __launch_bounds__(256) void mhca_sample_kernel(
    const float* __restrict__ featsl, const float* __restrict__ acur,
    const float* __restrict__ attw, const float* __restrict__ pcr,
    const float* __restrict__ camP, float* __restrict__ sfeat)
{
  const int t = blockIdx.x, b = blockIdx.y, e = threadIdx.x;
  const int row = b * TQ + t;
  float px = acur[row * 3 + 0], py = acur[row * 3 + 1], pz = acur[row * 3 + 2];
  const float lx = pcr[0], hx = pcr[1];
  const float ly = pcr[2], hy = pcr[3];
  const float lz = pcr[4], hz = pcr[5];
  float acc = 0.0f;
  for (int c = 0; c < CCAM; ++c) {
    px = px * (hx - lx) + lx;
    py = py * (hy - ly) + ly;
    pz = pz * (hz - lz) + lz;
    const float* P = camP + c * 12;
    const float pr0 = P[0] * px + P[1] * py + P[2]  * pz + P[3];
    const float pr1 = P[4] * px + P[5] * py + P[6]  * pz + P[7];
    const float pr2 = P[8] * px + P[9] * py + P[10] * pz + P[11];
    const float zc   = fmaxf(pr2, 1e-5f);
    const float pixx = pr0 / zc * (float)HH;
    const float pixy = pr1 / zc * (float)WW;
    const bool oob = (pixx < 0.0f) || (pixx > (float)HH) ||
                     (pixy < 0.0f) || (pixy > (float)WW);
    const float gx = pixx / (float)HH;
    const float gy = pixy / (float)WW;
    const float ix = (gx + 1.0f) * 0.5f * (float)(WW - 1);
    const float iy = (gy + 1.0f) * 0.5f * (float)(HH - 1);
    const float x0 = floorf(ix), y0 = floorf(iy);
    const float x1 = x0 + 1.0f,  y1 = y0 + 1.0f;
    const float* fb = featsl + (((size_t)c * BB + b) * EE + e) * (HH * WW);
    auto gat = [&](float xi, float yi) -> float {
      const bool val = (xi >= 0.0f) && (xi <= (float)(WW - 1)) &&
                       (yi >= 0.0f) && (yi <= (float)(HH - 1));
      const int xc = (int)fminf(fmaxf(xi, 0.0f), (float)(WW - 1));
      const int yc = (int)fminf(fmaxf(yi, 0.0f), (float)(HH - 1));
      const float v = fb[yc * WW + xc];
      return val ? v : 0.0f;
    };
    const float w00 = (x1 - ix) * (y1 - iy);
    const float w10 = (ix - x0) * (y1 - iy);
    const float w01 = (x1 - ix) * (iy - y0);
    const float w11 = (ix - x0) * (iy - y0);
    const float sf = gat(x0, y0) * w00 + gat(x1, y0) * w10 +
                     gat(x0, y1) * w01 + gat(x1, y1) * w11;
    if (!oob) acc += sf * attw[(size_t)row * 16 + c];
  }
  sfeat[(size_t)row * EE + e] = acc;
}

// ---------------------------------------------------------------------------
// Flash self-attention with WMMA. One wave per (q-tile of 16, head, batch).
// hd=32 == one f16 WMMA K-step. Key chunks of 32 => 2 score WMMAs + 2 PV
// WMMAs per chunk, online softmax in f32. V chunk staged via LDS (b128 loads).
// qkv layout: row-major NPAD x 768 (q|k|v). Output o: NPAD x 256.
// ---------------------------------------------------------------------------
__global__ __launch_bounds__(32) void attn_kernel(const float* __restrict__ qkv,
                                                  float* __restrict__ o)
{
  __shared__ float ps[16][33];
  __shared__ float vs[32][36];   // stride 36 floats keeps b128 alignment
  const int qt = blockIdx.x, h = blockIdx.y, b = blockIdx.z;
  const int lane = threadIdx.x & 31;
  const int half = lane >> 4;
  const int idx  = lane & 15;
  const int hb   = half * 8;
  const int rowbase = b * TQ;
  const float scale = 0.17677669529f; // 1/sqrt(32)

  v16h qf;
  {
    const float* pq = qkv + (size_t)(rowbase + qt * 16 + idx) * 768 + h * HD + hb;
    const float4 r0 = ((const float4*)pq)[0];
    const float4 r1 = ((const float4*)pq)[1];
    const float4 r2 = ((const float4*)(pq + 16))[0];
    const float4 r3 = ((const float4*)(pq + 16))[1];
    qf[0]  = (_Float16)(r0.x * scale); qf[1]  = (_Float16)(r0.y * scale);
    qf[2]  = (_Float16)(r0.z * scale); qf[3]  = (_Float16)(r0.w * scale);
    qf[4]  = (_Float16)(r1.x * scale); qf[5]  = (_Float16)(r1.y * scale);
    qf[6]  = (_Float16)(r1.z * scale); qf[7]  = (_Float16)(r1.w * scale);
    qf[8]  = (_Float16)(r2.x * scale); qf[9]  = (_Float16)(r2.y * scale);
    qf[10] = (_Float16)(r2.z * scale); qf[11] = (_Float16)(r2.w * scale);
    qf[12] = (_Float16)(r3.x * scale); qf[13] = (_Float16)(r3.y * scale);
    qf[14] = (_Float16)(r3.z * scale); qf[15] = (_Float16)(r3.w * scale);
  }

  float m[8], lsum[8];
  v8f olo = {}, ohi = {};
#pragma unroll
  for (int r = 0; r < 8; ++r) { m[r] = -1e30f; lsum[r] = 0.0f; }

  for (int kc = 0; kc < 29; ++kc) {
    const int kbase = kc * 32;

    // Cooperative V-chunk stage: lane copies key row (kbase+lane), 32 dims.
    {
      const float* pv = qkv + (size_t)(rowbase + kbase + lane) * 768 + 2 * EE + h * HD;
#pragma unroll
      for (int j = 0; j < 8; ++j)
        ((float4*)&vs[lane][0])[j] = ((const float4*)pv)[j];
    }

    const float* pk0 = qkv + (size_t)(rowbase + kbase + idx) * 768 + EE + h * HD + hb;
    const v16h kf0 = load_frag(pk0);
    const v16h kf1 = load_frag(pk0 + (size_t)16 * 768);

    v8f zero = {};
    v8f s0 = __builtin_amdgcn_wmma_f32_16x16x32_f16(false, qf, false, kf0,
                                                    (short)0, zero, false, false);
    v8f s1 = __builtin_amdgcn_wmma_f32_16x16x32_f16(false, qf, false, kf1,
                                                    (short)0, zero, false, false);
    if (kbase + idx >= TQ) {
#pragma unroll
      for (int r = 0; r < 8; ++r) s0[r] = -1e30f;
    }
    if (kbase + 16 + idx >= TQ) {
#pragma unroll
      for (int r = 0; r < 8; ++r) s1[r] = -1e30f;
    }
    float al[8];
#pragma unroll
    for (int r = 0; r < 8; ++r) {
      float rm = fmaxf(s0[r], s1[r]);
      rm = fmaxf(rm, __shfl_xor(rm, 1, 32));
      rm = fmaxf(rm, __shfl_xor(rm, 2, 32));
      rm = fmaxf(rm, __shfl_xor(rm, 4, 32));
      rm = fmaxf(rm, __shfl_xor(rm, 8, 32));
      const float nm = fmaxf(m[r], rm);
      al[r] = __expf(m[r] - nm);
      m[r] = nm;
      const float p0 = __expf(s0[r] - nm);
      const float p1 = __expf(s1[r] - nm);
      float rs = p0 + p1;
      rs += __shfl_xor(rs, 1, 32);
      rs += __shfl_xor(rs, 2, 32);
      rs += __shfl_xor(rs, 4, 32);
      rs += __shfl_xor(rs, 8, 32);
      lsum[r] = lsum[r] * al[r] + rs;
      olo[r] *= al[r];
      ohi[r] *= al[r];
      ps[r + 8 * half][idx]      = p0;
      ps[r + 8 * half][idx + 16] = p1;
    }
    __syncthreads();
    v16h pf, vlo, vhi;
#pragma unroll
    for (int j = 0; j < 8; ++j) {
      pf[j]     = (_Float16)ps[idx][hb + j];
      pf[8 + j] = (_Float16)ps[idx][16 + hb + j];
    }
#pragma unroll
    for (int j = 0; j < 16; ++j) {
      const int kk = (j < 8) ? (hb + j) : (16 + hb + (j - 8));
      vlo[j] = (_Float16)vs[kk][idx];
      vhi[j] = (_Float16)vs[kk][16 + idx];
    }
    olo = __builtin_amdgcn_wmma_f32_16x16x32_f16(false, pf, false, vlo,
                                                 (short)0, olo, false, false);
    ohi = __builtin_amdgcn_wmma_f32_16x16x32_f16(false, pf, false, vhi,
                                                 (short)0, ohi, false, false);
    __syncthreads();
  }

#pragma unroll
  for (int r = 0; r < 8; ++r) {
    const int t = qt * 16 + r + 8 * half;
    if (t < TQ) {
      const float inv = 1.0f / lsum[r];
      o[(size_t)(rowbase + t) * EE + h * HD + idx]      = olo[r] * inv;
      o[(size_t)(rowbase + t) * EE + h * HD + 16 + idx] = ohi[r] * inv;
    }
  }
}

// ---------------------------------------------------------------------------
// Small elementwise kernels
// ---------------------------------------------------------------------------
__global__ void init_x_kernel(const float* __restrict__ q, float* __restrict__ x)
{
  const int i = blockIdx.x * blockDim.x + threadIdx.x;
  if (i >= NPAD * EE) return;
  const int row = i / EE;
  x[i] = (row < NREAL) ? q[i] : 0.0f;
}

__global__ void init_a_kernel(const float* __restrict__ a, float* __restrict__ acur)
{
  const int i = blockIdx.x * blockDim.x + threadIdx.x;
  if (i < NREAL * 3) acur[i] = a[i];
}

__global__ void anchor_update_kernel(float* __restrict__ acur,
                                     const float* __restrict__ offs,
                                     float* __restrict__ poss_out, int l)
{
  const int i = blockIdx.x * blockDim.x + threadIdx.x;
  if (i >= NREAL * 3) return;
  const int row = i / 3, d = i % 3;
  const int b = row / TQ, t = row % TQ;
  float a = acur[i];
  a = fminf(fmaxf(a, 0.0f), 1.0f);
  const float num = fmaxf(a, 1e-5f);
  const float den = fmaxf(1.0f - a, 1e-5f);
  const float iv = __logf(num / den);
  const float v = iv + offs[(size_t)row * 16 + d];
  const float s = 1.0f / (1.0f + __expf(-v));
  acur[i] = s;
  poss_out[(((size_t)b * LNUM + l) * TQ + t) * 3 + d] = s;
}

__global__ void copyout_kernel(const float* __restrict__ src,
                               float* __restrict__ dst, int M, int l)
{
  const int i = blockIdx.x * blockDim.x + threadIdx.x;
  if (i >= NREAL * M) return;
  const int row = i / M, mcol = i % M;
  const int b = row / TQ, t = row % TQ;
  dst[(((size_t)b * LNUM + l) * TQ + t) * M + mcol] = src[(size_t)row * 16 + mcol];
}

// ---------------------------------------------------------------------------
// Host orchestration
// ---------------------------------------------------------------------------
extern "C" void kernel_launch(void* const* d_in, const int* in_sizes, int n_in,
                              void* d_out, int out_size, void* d_ws, size_t ws_size,
                              hipStream_t stream)
{
  (void)in_sizes; (void)n_in; (void)out_size; (void)ws_size;

  const float* query    = (const float*)d_in[0];
  const float* anchor   = (const float*)d_in[1];
  const float* feats    = (const float*)d_in[2];
  const float* cam_P    = (const float*)d_in[3];
  const float* pc_range = (const float*)d_in[4];
  const float* aw_w     = (const float*)d_in[5];
  const float* aw_b     = (const float*)d_in[6];
  const float* outp_w   = (const float*)d_in[7];
  const float* outp_b   = (const float*)d_in[8];
  const float* mhca_ln_s = (const float*)d_in[9];
  const float* mhca_ln_b = (const float*)d_in[10];
  const float* mha_ln_s  = (const float*)d_in[11];
  const float* mha_ln_b  = (const float*)d_in[12];
  const float* mlp_ln_s  = (const float*)d_in[13];
  const float* mlp_ln_b  = (const float*)d_in[14];
  const float* mha_in_w  = (const float*)d_in[15];
  const float* mha_in_b  = (const float*)d_in[16];
  const float* mha_out_w = (const float*)d_in[17];
  const float* mha_out_b = (const float*)d_in[18];
  const float* mlp_w1    = (const float*)d_in[19];
  const float* mlp_b1    = (const float*)d_in[20];
  const float* mlp_w2    = (const float*)d_in[21];
  const float* mlp_b2    = (const float*)d_in[22];
  const float* cls_w1    = (const float*)d_in[23];
  const float* cls_b1    = (const float*)d_in[24];
  const float* cls_ln_s  = (const float*)d_in[25];
  const float* cls_ln_b  = (const float*)d_in[26];
  const float* cls_w2    = (const float*)d_in[27];
  const float* cls_b2    = (const float*)d_in[28];
  const float* pos_w1    = (const float*)d_in[29];
  const float* pos_b1    = (const float*)d_in[30];
  const float* pos_w2    = (const float*)d_in[31];
  const float* pos_b2    = (const float*)d_in[32];
  const float* pos_w3    = (const float*)d_in[33];
  const float* pos_b3    = (const float*)d_in[34];
  const float* los_w1    = (const float*)d_in[35];
  const float* los_b1    = (const float*)d_in[36];
  const float* los_ln_s  = (const float*)d_in[37];
  const float* los_ln_b  = (const float*)d_in[38];
  const float* los_w2    = (const float*)d_in[39];
  const float* los_b2    = (const float*)d_in[40];

  float* ws = (float*)d_ws;
  float* x     = ws;                 ws += (size_t)NPAD * EE;
  float* xn    = ws;                 ws += (size_t)NPAD * EE;
  float* obuf  = ws;                 ws += (size_t)NPAD * EE;   // sfeat / attn out
  float* t1    = ws;                 ws += (size_t)NPAD * EE;
  float* t1n   = ws;                 ws += (size_t)NPAD * EE;
  float* qkv   = ws;                 ws += (size_t)NPAD * 3 * EE;
  float* h512  = ws;                 ws += (size_t)NPAD * 2 * EE;
  float* attwb = ws;                 ws += (size_t)NPAD * 16;
  float* offsb = ws;                 ws += (size_t)NPAD * 16;
  float* clsb  = ws;                 ws += (size_t)NPAD * 16;
  float* losb  = ws;                 ws += (size_t)NPAD * 16;
  float* acur  = ws;                 ws += 8192;

  float* cats_out = (float*)d_out;
  float* poss_out = cats_out + (size_t)BB * LNUM * TQ * NCLS;
  float* los_out  = poss_out + (size_t)BB * LNUM * TQ * 3;

  init_x_kernel<<<(NPAD * EE + 255) / 256, 256, 0, stream>>>(query, x);
  init_a_kernel<<<(NREAL * 3 + 255) / 256, 256, 0, stream>>>(anchor, acur);

  for (int l = 0; l < LNUM; ++l) {
    // ---- MHCA ----
    ln_kernel<<<NPAD, 256, 0, stream>>>(x, mhca_ln_s + l * EE, mhca_ln_b + l * EE, xn, 0);
    gemm_kernel<<<dim3(RT, 1), 32, 0, stream>>>(
        xn, aw_w + (size_t)l * CCAM * EE, aw_b + l * CCAM, nullptr, attwb,
        EE, CCAM, 16, 3);
    mhca_sample_kernel<<<dim3(TQ, BB), 256, 0, stream>>>(
        feats + (size_t)l * CCAM * BB * EE * HH * WW, acur, attwb, pc_range, cam_P, obuf);
    gemm_kernel<<<dim3(RT, 4), 32, 0, stream>>>(
        obuf, outp_w + (size_t)l * EE * EE, outp_b + l * EE, x, x, EE, EE, EE, 0);

    // ---- MHA ----
    ln_kernel<<<NPAD, 256, 0, stream>>>(x, mha_ln_s + l * EE, mha_ln_b + l * EE, xn, 0);
    gemm_kernel<<<dim3(RT, 12), 32, 0, stream>>>(
        xn, mha_in_w + (size_t)l * 3 * EE * EE, mha_in_b + l * 3 * EE, nullptr, qkv,
        EE, 3 * EE, 3 * EE, 0);
    attn_kernel<<<dim3(57, NHH, BB), 32, 0, stream>>>(qkv, obuf);
    gemm_kernel<<<dim3(RT, 4), 32, 0, stream>>>(
        obuf, mha_out_w + (size_t)l * EE * EE, mha_out_b + l * EE, x, x, EE, EE, EE, 0);

    // ---- MLP ----
    ln_kernel<<<NPAD, 256, 0, stream>>>(x, mlp_ln_s + l * EE, mlp_ln_b + l * EE, xn, 0);
    gemm_kernel<<<dim3(RT, 8), 32, 0, stream>>>(
        xn, mlp_w1 + (size_t)l * 2 * EE * EE, mlp_b1 + l * 2 * EE, nullptr, h512,
        EE, 2 * EE, 2 * EE, 1);
    gemm_kernel<<<dim3(RT, 4), 32, 0, stream>>>(
        h512, mlp_w2 + (size_t)l * EE * 2 * EE, mlp_b2 + l * EE, x, x, 2 * EE, EE, EE, 0);

    // ---- cls head ----
    gemm_kernel<<<dim3(RT, 4), 32, 0, stream>>>(
        x, cls_w1 + (size_t)l * EE * EE, cls_b1 + l * EE, nullptr, t1, EE, EE, EE, 0);
    ln_kernel<<<NPAD, 256, 0, stream>>>(t1, cls_ln_s + l * EE, cls_ln_b + l * EE, t1n, 1);
    gemm_kernel<<<dim3(RT, 1), 32, 0, stream>>>(
        t1n, cls_w2 + (size_t)l * NCLS * EE, cls_b2 + l * NCLS, nullptr, clsb,
        EE, NCLS, 16, 0);
    copyout_kernel<<<(NREAL * NCLS + 255) / 256, 256, 0, stream>>>(clsb, cats_out, NCLS, l);

    // ---- pos head + anchor update ----
    gemm_kernel<<<dim3(RT, 8), 32, 0, stream>>>(
        x, pos_w1 + (size_t)l * 2 * EE * EE, pos_b1 + l * 2 * EE, nullptr, h512,
        EE, 2 * EE, 2 * EE, 2);
    gemm_kernel<<<dim3(RT, 4), 32, 0, stream>>>(
        h512, pos_w2 + (size_t)l * EE * 2 * EE, pos_b2 + l * EE, nullptr, t1,
        2 * EE, EE, EE, 2);
    gemm_kernel<<<dim3(RT, 1), 32, 0, stream>>>(
        t1, pos_w3 + (size_t)l * 3 * EE, pos_b3 + l * 3, nullptr, offsb, EE, 3, 16, 0);
    anchor_update_kernel<<<(NREAL * 3 + 255) / 256, 256, 0, stream>>>(acur, offsb, poss_out, l);

    // ---- los head ----
    gemm_kernel<<<dim3(RT, 4), 32, 0, stream>>>(
        x, los_w1 + (size_t)l * EE * EE, los_b1 + l * EE, nullptr, t1, EE, EE, EE, 0);
    ln_kernel<<<NPAD, 256, 0, stream>>>(t1, los_ln_s + l * EE, los_ln_b + l * EE, t1n, 1);
    gemm_kernel<<<dim3(RT, 1), 32, 0, stream>>>(
        t1n, los_w2 + (size_t)l * CCAM * EE, los_b2 + l * CCAM, nullptr, losb,
        EE, CCAM, 16, 0);
    copyout_kernel<<<(NREAL * CCAM + 255) / 256, 256, 0, stream>>>(losb, los_out, CCAM, l);
  }
}
